// SparseMoELayer_44083544326409
// MI455X (gfx1250) — matrix operand order repaired and checked by reading, not compiled, not run on hardware
//
#include <hip/hip_runtime.h>
#include <hip/hip_bf16.h>
#include <math.h>

// ---------------------------------------------------------------------------
// Sparse MoE (top-2 of 8 experts), d_model=1024, d_ff=4096, N=4096 tokens.
// fp32 inputs; GEMMs run on v_wmma_f32_16x16x32_bf16 (fp32 accumulate) with
// in-register fp32->bf16 packing while staging tiles through LDS.
// ---------------------------------------------------------------------------

#define E_NUM   8
#define CDIM    1024
#define FDIM    4096
#define NTOK    4096
#define LAMBDA  0.01f

typedef __attribute__((ext_vector_type(16))) __bf16 v16bf;
typedef __attribute__((ext_vector_type(8)))  __bf16 v8bf;
typedef __attribute__((ext_vector_type(8)))  float  v8f;

// ---- workspace layout (bytes) ----
// [0,64)      int   cnt[8]        per-expert routed-token count
// [64,128)    float probsum[8]    per-expert sum of softmax probs (for I_i)
// [128,192)   int   base[8]       per-expert h-pool row offset (64-padded)
// [1024, +128KB)   int   list[8*4096]   gathered token ids
// [+128KB, +256KB) float wlist[8*4096]  gathered combine weights
// [WS_H, ...) bf16  h[<=8704 rows][4096] expert hidden activations (~71 MB)
#define WS_LIST   1024
#define WS_WLIST  (WS_LIST + E_NUM * NTOK * 4)
#define WS_H      (WS_WLIST + E_NUM * NTOK * 4)

// ---------------------------------------------------------------------------
__device__ __forceinline__ unsigned pack_bf16x2(float a, float b) {
    const __bf16 lo = (__bf16)a;           // RNE fp32->bf16
    const __bf16 hi = (__bf16)b;
    return ((unsigned)__builtin_bit_cast(unsigned short, hi) << 16) |
           (unsigned)__builtin_bit_cast(unsigned short, lo);
}

// ---------------------------------------------------------------------------
__global__ void moe_zero(float* __restrict__ out, int n, int* __restrict__ hdr) {
    int i = blockIdx.x * blockDim.x + threadIdx.x;
    if (i < n) out[i] = 0.0f;
    if (i < 64) hdr[i] = 0;   // zero cnt[], probsum[], base[]
}

// ---------------------------------------------------------------------------
// One wave32 per token: logits -> softmax -> top-2 -> gather lists + stats.
__global__ __launch_bounds__(256) void moe_gate(
    const float* __restrict__ x, const float* __restrict__ gw,
    int* __restrict__ cnt, float* __restrict__ probsum,
    int* __restrict__ list, float* __restrict__ wlist) {
    const int lane = threadIdx.x & 31;
    const int t = blockIdx.x * 8 + (threadIdx.x >> 5);

    float acc[E_NUM];
#pragma unroll
    for (int e = 0; e < E_NUM; ++e) acc[e] = 0.0f;

    const float* xr = x + (size_t)t * CDIM;
    for (int c = lane; c < CDIM; c += 32) {
        const float xv = xr[c];
        const float* g = gw + (size_t)c * E_NUM;
#pragma unroll
        for (int e = 0; e < E_NUM; ++e) acc[e] = fmaf(xv, g[e], acc[e]);
    }
#pragma unroll
    for (int off = 16; off >= 1; off >>= 1) {
#pragma unroll
        for (int e = 0; e < E_NUM; ++e) acc[e] += __shfl_xor(acc[e], off, 32);
    }

    if (lane == 0) {
        float mx = acc[0];
#pragma unroll
        for (int e = 1; e < E_NUM; ++e) mx = fmaxf(mx, acc[e]);
        float p[E_NUM];
        float s = 0.0f;
#pragma unroll
        for (int e = 0; e < E_NUM; ++e) { p[e] = __expf(acc[e] - mx); s += p[e]; }
        const float inv = 1.0f / s;
#pragma unroll
        for (int e = 0; e < E_NUM; ++e) {
            p[e] *= inv;
            atomicAdd(&probsum[e], p[e]);
        }
        // top-2 (first-index wins ties, matching lax.top_k)
        int i0 = 0;
#pragma unroll
        for (int e = 1; e < E_NUM; ++e) if (p[e] > p[i0]) i0 = e;
        int i1 = (i0 == 0) ? 1 : 0;
#pragma unroll
        for (int e = 0; e < E_NUM; ++e) if (e != i0 && p[e] > p[i1]) i1 = e;
        const float wsum = p[i0] + p[i1];
        const float w0 = p[i0] / wsum, w1v = p[i1] / wsum;
        int pos0 = atomicAdd(&cnt[i0], 1);
        list[i0 * NTOK + pos0] = t; wlist[i0 * NTOK + pos0] = w0;
        int pos1 = atomicAdd(&cnt[i1], 1);
        list[i1 * NTOK + pos1] = t; wlist[i1 * NTOK + pos1] = w1v;
    }
}

// ---------------------------------------------------------------------------
__global__ void moe_prefix(const int* __restrict__ cnt, int* __restrict__ base) {
    if (threadIdx.x == 0 && blockIdx.x == 0) {
        int tot = 0;
        for (int e = 0; e < E_NUM; ++e) {
            base[e] = tot;
            tot += ((cnt[e] + 63) >> 6) << 6;   // pad each expert region to 64 rows
        }
    }
}

// ---------------------------------------------------------------------------
// CDNA5 16-bit WMMA fragment layouts (ISA 7.12.2, wave32):
//   A (16x32):  lane m=lane%16 holds K runs [hi*8,+8) and [16+hi*8,+8), hi=lane/16
//   B (32x16):  lane n=lane%16 holds K runs [hi*8,+8) and [16+hi*8,+8), same as A
// LDS tile rows padded to 72 elements (144B): fragment b128 loads stay
// 16B-aligned and the 16 lanes of a half-wave hit distinct bank groups.
#define KTILE 64
#define LDROW 72

union frag16 { v16bf v; v8bf h8[2]; };

// ---------------------------------------------------------------------------
// Stage 1: h = gelu(x_gathered @ w1[e] + b1[e]) stored as bf16.
// Block: 64 gathered rows x 128 F-cols per sweep; 8 waves x 4 row-tiles,
// K staged 64 at a time (8 v_wmma per wave per k-step).
__global__ __launch_bounds__(256) void moe_ffn1(
    const float* __restrict__ x, const float* __restrict__ w1,
    const float* __restrict__ b1, const int* __restrict__ cnt,
    const int* __restrict__ base, const int* __restrict__ list,
    __bf16* __restrict__ h) {
    const int e = blockIdx.y;
    const int ce = cnt[e];
    const int row0 = blockIdx.x * 64;
    if (row0 >= ce) return;

    __shared__ __bf16 As[64][LDROW];
    __shared__ __bf16 Bs[128][LDROW];
    __shared__ int toks[64];

    const int tid = threadIdx.x;
    if (tid < 64) {
        const int r = row0 + tid;
        toks[tid] = list[e * NTOK + (r < ce ? r : 0)];
    }
    __syncthreads();

    const float* w1e = w1 + (size_t)e * CDIM * FDIM;
    const float* b1e = b1 + (size_t)e * FDIM;
    const int lane = tid & 31, wv = tid >> 5;
    const int mr = lane & 15, hi = lane >> 4;
    const int nb = wv * 16 + mr;
    const size_t hrow0 = (size_t)(base[e] + row0);

    for (int fc = 0; fc < FDIM / 128; ++fc) {
        const int f0 = fc * 128;
        v8f acc[4];
        const v8f zero = {};
#pragma unroll
        for (int rt = 0; rt < 4; ++rt) acc[rt] = zero;

        for (int kt = 0; kt < CDIM / KTILE; ++kt) {
            const int k0 = kt * KTILE;
            // A tile: 64 rows x 64 K; 8 fp32 -> one packed ds_store_b128
#pragma unroll
            for (int i = 0; i < 2; ++i) {
                const int u = tid + i * 256;
                const int am = u >> 3, ak8 = (u & 7) * 8;
                const float* src = x + (size_t)toks[am] * CDIM + (k0 + ak8);
                const float4 a0 = *(const float4*)(src);
                const float4 a1 = *(const float4*)(src + 4);
                uint4 p;
                p.x = pack_bf16x2(a0.x, a0.y);
                p.y = pack_bf16x2(a0.z, a0.w);
                p.z = pack_bf16x2(a1.x, a1.y);
                p.w = pack_bf16x2(a1.z, a1.w);
                *(uint4*)&As[am][ak8] = p;
            }
            // B tile (transposed): 128 cols x 64 K; 4 coalesced rows -> ds_store_b64
#pragma unroll
            for (int i = 0; i < 8; ++i) {
                const int u = tid + i * 256;
                const int nn = u & 127, kk4 = (u >> 7) * 4;
                const float* src = w1e + (size_t)(k0 + kk4) * FDIM + (f0 + nn);
                uint2 p;
                p.x = pack_bf16x2(src[0], src[FDIM]);
                p.y = pack_bf16x2(src[2 * FDIM], src[3 * FDIM]);
                *(uint2*)&Bs[nn][kk4] = p;
            }
            // prefetch next weight k-slab (global_prefetch_b8)
            if (kt + 1 < CDIM / KTILE) {
                const int pk = k0 + KTILE + (tid >> 2);
                __builtin_prefetch(w1e + (size_t)pk * FDIM + f0 + (tid & 3) * 32, 0, 1);
            }
            __syncthreads();

#pragma unroll
            for (int ks = 0; ks < KTILE; ks += 32) {
                frag16 bf, af;
                bf.h8[0] = *(const v8bf*)&Bs[nb][ks + hi * 8];
                bf.h8[1] = *(const v8bf*)&Bs[nb][ks + 16 + hi * 8];
#pragma unroll
                for (int rt = 0; rt < 4; ++rt) {
                    af.h8[0] = *(const v8bf*)&As[rt * 16 + mr][ks + hi * 8];
                    af.h8[1] = *(const v8bf*)&As[rt * 16 + mr][ks + 16 + hi * 8];
                    acc[rt] = __builtin_amdgcn_wmma_f32_16x16x32_bf16(
                        false, af.v, false, bf.v, (short)0, acc[rt], false, false);
                }
            }
            __syncthreads();
        }

        // epilogue: +b1, exact GELU, store bf16
        const int col = f0 + wv * 16 + mr;
        const float bias = b1e[col];
#pragma unroll
        for (int rt = 0; rt < 4; ++rt) {
#pragma unroll
            for (int r = 0; r < 8; ++r) {
                float v = acc[rt][r] + bias;
                v = 0.5f * v * (1.0f + erff(v * 0.70710678118654752f));
                h[(hrow0 + (size_t)(rt * 16 + r + hi * 8)) * FDIM + col] = (__bf16)v;
            }
        }
    }
}

// ---------------------------------------------------------------------------
// Stage 2: out[token] += weight * (h @ w2[e] + b2[e])  (atomic scatter-add)
__global__ __launch_bounds__(256) void moe_ffn2(
    const __bf16* __restrict__ h, const float* __restrict__ w2,
    const float* __restrict__ b2, const int* __restrict__ cnt,
    const int* __restrict__ base, const int* __restrict__ list,
    const float* __restrict__ wlist, float* __restrict__ out) {
    const int e = blockIdx.y;
    const int ce = cnt[e];
    const int row0 = blockIdx.x * 64;
    if (row0 >= ce) return;

    __shared__ __bf16 As[64][LDROW];
    __shared__ __bf16 Bs[128][LDROW];
    __shared__ int toks[64];
    __shared__ float twgt[64];

    const int tid = threadIdx.x;
    if (tid < 64) {
        const int r = row0 + tid;
        const bool ok = r < ce;
        toks[tid] = list[e * NTOK + (ok ? r : 0)];
        twgt[tid] = ok ? wlist[e * NTOK + r] : 0.0f;
    }
    __syncthreads();

    const float* w2e = w2 + (size_t)e * FDIM * CDIM;
    const float* b2e = b2 + (size_t)e * CDIM;
    const int lane = tid & 31, wv = tid >> 5;
    const int mr = lane & 15, hi = lane >> 4;
    const int nb = wv * 16 + mr;
    const size_t hrow0 = (size_t)(base[e] + row0);

    for (int fc = 0; fc < CDIM / 128; ++fc) {
        const int f0 = fc * 128;
        v8f acc[4];
        const v8f zero = {};
#pragma unroll
        for (int rt = 0; rt < 4; ++rt) acc[rt] = zero;

        for (int kt = 0; kt < FDIM / KTILE; ++kt) {
            const int k0 = kt * KTILE;
            // A tile: h rows already bf16 -> straight b128 copies
#pragma unroll
            for (int i = 0; i < 2; ++i) {
                const int u = tid + i * 256;
                const int am = u >> 3, ak8 = (u & 7) * 8;
                *(uint4*)&As[am][ak8] =
                    *(const uint4*)&h[(hrow0 + am) * FDIM + (k0 + ak8)];
            }
            // B tile (transposed): 128 cols x 64 K from w2
#pragma unroll
            for (int i = 0; i < 8; ++i) {
                const int u = tid + i * 256;
                const int nn = u & 127, kk4 = (u >> 7) * 4;
                const float* src = w2e + (size_t)(k0 + kk4) * CDIM + (f0 + nn);
                uint2 p;
                p.x = pack_bf16x2(src[0], src[CDIM]);
                p.y = pack_bf16x2(src[2 * CDIM], src[3 * CDIM]);
                *(uint2*)&Bs[nn][kk4] = p;
            }
            if (kt + 1 < FDIM / KTILE) {
                const int pk = k0 + KTILE + (tid >> 2);
                __builtin_prefetch(w2e + (size_t)pk * CDIM + f0 + (tid & 3) * 32, 0, 1);
            }
            __syncthreads();

#pragma unroll
            for (int ks = 0; ks < KTILE; ks += 32) {
                frag16 bf, af;
                bf.h8[0] = *(const v8bf*)&Bs[nb][ks + hi * 8];
                bf.h8[1] = *(const v8bf*)&Bs[nb][ks + 16 + hi * 8];
#pragma unroll
                for (int rt = 0; rt < 4; ++rt) {
                    af.h8[0] = *(const v8bf*)&As[rt * 16 + mr][ks + hi * 8];
                    af.h8[1] = *(const v8bf*)&As[rt * 16 + mr][ks + 16 + hi * 8];
                    acc[rt] = __builtin_amdgcn_wmma_f32_16x16x32_bf16(
                        false, af.v, false, bf.v, (short)0, acc[rt], false, false);
                }
            }
            __syncthreads();
        }

        const int col = f0 + wv * 16 + mr;
        const float bias = b2e[col];
#pragma unroll
        for (int rt = 0; rt < 4; ++rt) {
#pragma unroll
            for (int r = 0; r < 8; ++r) {
                const int lr = rt * 16 + r + hi * 8;
                if (row0 + lr < ce) {
                    const float y = (acc[rt][r] + bias) * twgt[lr];
                    atomicAdd(&out[(size_t)toks[lr] * CDIM + col], y);
                }
            }
        }
    }
}

// ---------------------------------------------------------------------------
__global__ void moe_loss(const int* __restrict__ cnt,
                         const float* __restrict__ probsum,
                         float* __restrict__ out_loss) {
    if (threadIdx.x != 0 || blockIdx.x != 0) return;
    float f[E_NUM], I[E_NUM];
    float tot = 0.0f;
    for (int e = 0; e < E_NUM; ++e) tot += (float)cnt[e];
    const float invt = 1.0f / tot;
    float mf = 0.0f, mi = 0.0f;
    for (int e = 0; e < E_NUM; ++e) {
        f[e] = (float)cnt[e] * invt;
        I[e] = probsum[e] * (1.0f / (float)NTOK);
        mf += f[e]; mi += I[e];
    }
    mf *= (1.0f / E_NUM); mi *= (1.0f / E_NUM);
    float vf = 0.0f, vi = 0.0f;
    for (int e = 0; e < E_NUM; ++e) {
        const float df = f[e] - mf, di = I[e] - mi;
        vf += df * df; vi += di * di;
    }
    vf *= (1.0f / (E_NUM - 1));   // unbiased (ddof=1)
    vi *= (1.0f / (E_NUM - 1));
    out_loss[0] = LAMBDA * vf + LAMBDA * vi;
}

// ---------------------------------------------------------------------------
extern "C" void kernel_launch(void* const* d_in, const int* in_sizes, int n_in,
                              void* d_out, int out_size, void* d_ws, size_t ws_size,
                              hipStream_t stream) {
    (void)in_sizes; (void)n_in; (void)ws_size;
    const float* x  = (const float*)d_in[0];
    const float* gw = (const float*)d_in[1];
    const float* w1 = (const float*)d_in[2];
    const float* b1 = (const float*)d_in[3];
    const float* w2 = (const float*)d_in[4];
    const float* b2 = (const float*)d_in[5];
    float* out = (float*)d_out;

    char* ws = (char*)d_ws;
    int*    cnt     = (int*)(ws + 0);
    float*  probsum = (float*)(ws + 64);
    int*    base    = (int*)(ws + 128);
    int*    list    = (int*)(ws + WS_LIST);
    float*  wlist   = (float*)(ws + WS_WLIST);
    __bf16* h       = (__bf16*)(ws + WS_H);

    // 1) zero output + counters (harness poisons buffers; must re-init per call)
    const int zblocks = (out_size + 255) / 256;
    moe_zero<<<zblocks, 256, 0, stream>>>(out, out_size, (int*)ws);

    // 2) gating: softmax + top-2 + gather lists
    moe_gate<<<NTOK / 8, 256, 0, stream>>>(x, gw, cnt, probsum, list, wlist);

    // 3) per-expert h-pool offsets
    moe_prefix<<<1, 32, 0, stream>>>(cnt, base);

    // 4) expert FFN up-projection + GELU (bf16 WMMA)
    moe_ffn1<<<dim3(NTOK / 64, E_NUM), 256, 0, stream>>>(x, w1, b1, cnt, base, list, h);

    // 5) expert FFN down-projection + weighted scatter-add (bf16 WMMA)
    moe_ffn2<<<dim3(NTOK / 64, E_NUM), 256, 0, stream>>>(h, w2, b2, cnt, base, list, wlist, out);

    // 6) load-balance loss scalar
    moe_loss<<<1, 32, 0, stream>>>(cnt, probsum, out + (size_t)NTOK * CDIM);
}
